// RouterCond_14937896256004
// MI455X (gfx1250) — matrix operand order repaired and checked by reading, not compiled, not run on hardware
//
#include <hip/hip_runtime.h>
#include <math.h>

// ---------------------------------------------------------------------------
// RouterCond fused MLP router for MI455X (gfx1250, wave32, WMMA)
//   logits = GELU(concat(x,c) @ W1 + b1) @ W2 + b2 ; softmax ; top-2 ; norm
// bf16 operands, f32 WMMA accumulation (v_wmma_f32_16x16x32_bf16).
// W1/W2 pre-converted to transposed bf16 in d_ws; per-F-block W1 slices are
// staged with async Global->LDS copies (ASYNCcnt) into a ping-pong buffer.
// ---------------------------------------------------------------------------

typedef __attribute__((ext_vector_type(16))) __bf16 v16bf;
typedef __attribute__((ext_vector_type(8)))  __bf16 v8bf;
typedef __attribute__((ext_vector_type(8)))  float  v8f;
typedef __attribute__((ext_vector_type(4)))  int    v4i_t;

#define N_TOK   131072      // 16 * 8192
#define HDIM    256
#define CDIM    256
#define DDIM    512         // HDIM + CDIM
#define FDIM    1024
#define EDIM    16
#define MT      128         // tokens per block (8 waves x 16-row m-tiles)

// LDS strides padded for bank-conflict-free, 16B-aligned fragment loads
#define XLD     520         // 512 + 8  (1040B stride: +4 banks/row, 16B aligned)
#define WLD     520         // WbT rows (K dim 512 + 8)
#define W2LD    1032        // 1024 + 8 (2064B stride)
#define SCRLD   72          // 32 + 40  (144B stride: distinct banks over 16 rows)

#if __has_builtin(__builtin_amdgcn_global_load_async_to_lds_b128) && \
    __has_builtin(__builtin_amdgcn_s_wait_asynccnt)
#define HAVE_ASYNC_LDS 1
#else
#define HAVE_ASYNC_LDS 0
#endif

__device__ __forceinline__ float gelu_exact(float x) {
    return 0.5f * x * (1.0f + erff(x * 0.70710678118654752f));
}

// ---- pre-pass: W1 (512x1024 f32) -> W1T (1024x512 bf16, transposed),
//                W2 (1024x16  f32) -> W2T (16x1024  bf16, transposed) ----
__global__ __launch_bounds__(256)
void convert_weights_kernel(const float* __restrict__ W1,
                            const float* __restrict__ W2,
                            __bf16* __restrict__ W1T,
                            __bf16* __restrict__ W2T)
{
    int idx = blockIdx.x * 256 + threadIdx.x;
    if (idx < FDIM * DDIM) {
        int f = idx >> 9;            // / DDIM
        int k = idx & (DDIM - 1);
        W1T[idx] = (__bf16)W1[(size_t)k * FDIM + f];
    } else {
        int j = idx - FDIM * DDIM;
        if (j < EDIM * FDIM) {
            int e = j >> 10;         // / FDIM
            int k = j & (FDIM - 1);
            W2T[j] = (__bf16)W2[(size_t)k * EDIM + e];
        }
    }
}

__global__ __launch_bounds__(256, 1)
void router_fused_kernel(const float* __restrict__ inputs,
                         const float* __restrict__ cond,
                         const float* __restrict__ W1,
                         const float* __restrict__ b1,
                         const float* __restrict__ W2,
                         const float* __restrict__ b2,
                         const __bf16* __restrict__ W1Tg,
                         const __bf16* __restrict__ W2Tg,
                         int useWs,
                         float* __restrict__ out)
{
    __shared__ __align__(16) __bf16 Xs[MT * XLD];         // 133120 B
    __shared__ __align__(16) __bf16 WbT[2][32 * WLD];     //  66560 B (ping-pong)
    __shared__ __align__(16) __bf16 W2s[EDIM * W2LD];     //  33024 B
    __shared__ __align__(16) __bf16 scr[8 * 16 * SCRLD];  //  18432 B
    __shared__ float b1s[FDIM];
    __shared__ float b2s[EDIM];
    __shared__ float logitsS[MT * EDIM];

    const int tid  = threadIdx.x;
    const int w    = tid >> 5;       // wave id 0..7 == m-tile
    const int lane = tid & 31;
    const int lm   = lane & 15;
    const int lh   = lane >> 4;
    const int tok0 = blockIdx.x * MT;

    // stage one 32-row slice of W1 (bf16, transposed layout) into buffer `buf`
    auto stageW1 = [&](int buf, int fb) {
        if (useWs) {
            const __bf16* src = W1Tg + (size_t)fb * 32 * DDIM;   // 32 rows x 512
            for (int i = tid; i < 32 * 64; i += 256) {           // 16B chunks
                int c = i >> 6, j = i & 63;
                const __bf16* g = src + c * DDIM + j * 8;
                __bf16* l = &WbT[buf][c * WLD + j * 8];
#if HAVE_ASYNC_LDS
                __builtin_amdgcn_global_load_async_to_lds_b128(
                    (__attribute__((address_space(1))) v4i_t*)g,
                    (__attribute__((address_space(3))) v4i_t*)l, 0, 0);
#else
                *(uint4*)l = *(const uint4*)g;
#endif
            }
        } else {
            // fallback: transpose-convert from fp32 W1
            for (int i = tid; i < DDIM * 8; i += 256) {
                int k  = i >> 3;
                int c4 = (i & 7) << 2;
                float4 vw = *(const float4*)(W1 + (size_t)k * FDIM + fb * 32 + c4);
                WbT[buf][(c4 + 0) * WLD + k] = (__bf16)vw.x;
                WbT[buf][(c4 + 1) * WLD + k] = (__bf16)vw.y;
                WbT[buf][(c4 + 2) * WLD + k] = (__bf16)vw.z;
                WbT[buf][(c4 + 3) * WLD + k] = (__bf16)vw.w;
            }
        }
    };

    // ---- prefetch first W1 slice (overlaps with X/W2 staging below) ----
    stageW1(0, 0);

    // ---- stage X tile: concat(inputs, cond) -> bf16 LDS ----
    for (int i = tid; i < MT * (HDIM / 4); i += 256) {
        int r  = i >> 6;
        int c4 = (i & 63) << 2;
        size_t t = (size_t)(tok0 + r);
        float4 vi = *(const float4*)(inputs + t * HDIM + c4);
        float4 vc = *(const float4*)(cond   + t * CDIM + c4);
        __bf16* xr = &Xs[r * XLD];
        xr[c4 + 0] = (__bf16)vi.x; xr[c4 + 1] = (__bf16)vi.y;
        xr[c4 + 2] = (__bf16)vi.z; xr[c4 + 3] = (__bf16)vi.w;
        xr[HDIM + c4 + 0] = (__bf16)vc.x; xr[HDIM + c4 + 1] = (__bf16)vc.y;
        xr[HDIM + c4 + 2] = (__bf16)vc.z; xr[HDIM + c4 + 3] = (__bf16)vc.w;
    }
    // ---- stage W2 (transposed bf16) into LDS ----
    if (useWs) {
        for (int i = tid; i < EDIM * 128; i += 256) {    // 16 rows x 128 16B chunks
            int r = i >> 7, j = i & 127;
            *(uint4*)&W2s[r * W2LD + j * 8] = *(const uint4*)(W2Tg + r * FDIM + j * 8);
        }
    } else {
        for (int i = tid; i < FDIM * (EDIM / 4); i += 256) {
            int k  = i >> 2;
            int c4 = (i & 3) << 2;
            float4 vw = *(const float4*)(W2 + (size_t)k * EDIM + c4);
            W2s[(c4 + 0) * W2LD + k] = (__bf16)vw.x;
            W2s[(c4 + 1) * W2LD + k] = (__bf16)vw.y;
            W2s[(c4 + 2) * W2LD + k] = (__bf16)vw.z;
            W2s[(c4 + 3) * W2LD + k] = (__bf16)vw.w;
        }
    }
    for (int i = tid; i < FDIM; i += 256) b1s[i] = b1[i];
    if (tid < EDIM) b2s[tid] = b2[tid];

    v8f lacc;
    #pragma unroll
    for (int i = 0; i < 8; ++i) lacc[i] = 0.0f;

    const __bf16* xrow = &Xs[(w * 16 + lm) * XLD];
    __bf16* sc = &scr[w * 16 * SCRLD];

    for (int fb = 0; fb < FDIM / 32; ++fb) {
#if HAVE_ASYNC_LDS
        if (useWs) __builtin_amdgcn_s_wait_asynccnt(0);
#endif
        __syncthreads();                 // slice fb ready; prev reads of other buf done
        if (fb + 1 < FDIM / 32) stageW1((fb + 1) & 1, fb + 1);  // overlap with compute

        const __bf16* wr0 = &WbT[fb & 1][lm * WLD];
        const __bf16* wr1 = &WbT[fb & 1][(16 + lm) * WLD];

        // ---- GEMM1: two 16x16 f-tiles for m-tile w, K = 512 (16 WMMA steps) ----
        float bv0 = b1s[fb * 32 + lm];
        float bv1 = b1s[fb * 32 + 16 + lm];
        v8f acc0, acc1;
        #pragma unroll
        for (int i = 0; i < 8; ++i) { acc0[i] = bv0; acc1[i] = bv1; }

        #pragma unroll 4
        for (int k16 = 0; k16 < DDIM / 32; ++k16) {
            const int kb = k16 * 32;
            v8bf alo = *(const v8bf*)(xrow + kb + lh * 8);
            v8bf ahi = *(const v8bf*)(xrow + kb + 16 + lh * 8);
            v16bf a;
            #pragma unroll
            for (int i = 0; i < 8; ++i) { a[i] = alo[i]; a[8 + i] = ahi[i]; }
            v8bf p0 = *(const v8bf*)(wr0 + kb + lh * 8);
            v8bf p1 = *(const v8bf*)(wr0 + kb + 16 + lh * 8);
            v8bf q0 = *(const v8bf*)(wr1 + kb + lh * 8);
            v8bf q1 = *(const v8bf*)(wr1 + kb + 16 + lh * 8);
            v16bf bf0, bf1;
            #pragma unroll
            for (int i = 0; i < 8; ++i) {
                bf0[i] = p0[i]; bf0[8 + i] = p1[i];
                bf1[i] = q0[i]; bf1[8 + i] = q1[i];
            }
            acc0 = __builtin_amdgcn_wmma_f32_16x16x32_bf16(
                       false, a, false, bf0, (short)0, acc0, false, false);
            acc1 = __builtin_amdgcn_wmma_f32_16x16x32_bf16(
                       false, a, false, bf1, (short)0, acc1, false, false);
        }

        // ---- GELU, re-layout D->A via per-wave LDS scratch ----
        #pragma unroll
        for (int v = 0; v < 8; ++v) {
            int mloc = v + lh * 8;
            sc[mloc * SCRLD + lm]      = (__bf16)gelu_exact(acc0[v]);
            sc[mloc * SCRLD + 16 + lm] = (__bf16)gelu_exact(acc1[v]);
        }
        __builtin_amdgcn_wave_barrier();  // same-wave LDS ops are in-order

        // ---- GEMM2 partial: logits += h_tile(16x32) @ W2[fb*32..][16] ----
        v8bf a2lo = *(const v8bf*)(sc + lm * SCRLD + lh * 8);
        v8bf a2hi = *(const v8bf*)(sc + lm * SCRLD + 16 + lh * 8);
        v16bf a2;
        #pragma unroll
        for (int i = 0; i < 8; ++i) { a2[i] = a2lo[i]; a2[8 + i] = a2hi[i]; }
        const __bf16* w2r = &W2s[lm * W2LD + fb * 32 + lh * 8];
        v8bf b2lo = *(const v8bf*)(w2r);
        v8bf b2hi = *(const v8bf*)(w2r + 16);
        v16bf b2f;
        #pragma unroll
        for (int i = 0; i < 8; ++i) { b2f[i] = b2lo[i]; b2f[8 + i] = b2hi[i]; }
        lacc = __builtin_amdgcn_wmma_f32_16x16x32_bf16(
                   false, a2, false, b2f, (short)0, lacc, false, false);
    }

    // ---- logits (+b2) to LDS ----
    #pragma unroll
    for (int v = 0; v < 8; ++v)
        logitsS[(w * 16 + lh * 8 + v) * EDIM + lm] = lacc[v] + b2s[lm];
    __syncthreads();

    // ---- softmax / clamp / top-2 / normalize / writeback ----
    if (tid < MT) {
        const float* lr = &logitsS[tid * EDIM];
        float p[EDIM];
        float mx = lr[0];
        #pragma unroll
        for (int e = 1; e < EDIM; ++e) mx = fmaxf(mx, lr[e]);
        float s = 0.0f;
        #pragma unroll
        for (int e = 0; e < EDIM; ++e) { p[e] = __expf(lr[e] - mx); s += p[e]; }
        float inv = 1.0f / s;
        #pragma unroll
        for (int e = 0; e < EDIM; ++e) {
            float q = p[e] * inv;
            p[e] = fminf(fmaxf(q, 1e-9f), 1.0f - 1e-9f);
        }
        int i1 = 0; float v1 = p[0];
        #pragma unroll
        for (int e = 1; e < EDIM; ++e) if (p[e] > v1) { v1 = p[e]; i1 = e; }
        int i2 = (i1 == 0) ? 1 : 0; float v2 = p[i2];
        #pragma unroll
        for (int e = 0; e < EDIM; ++e)
            if (e != i1 && p[e] > v2) { v2 = p[e]; i2 = e; }
        float rs = 1.0f / (v1 + v2);

        size_t t = (size_t)tok0 + tid;
        float* om  = out + t * EDIM;                        // router_mask
        float* oi  = out + (size_t)N_TOK * 16 + t * 2;      // top_idx (as float)
        float* orp = out + (size_t)N_TOK * 18 + t * EDIM;   // router_probs
        float* opf = out + (size_t)N_TOK * 34 + t * EDIM;   // probs_full
        #pragma unroll
        for (int e = 0; e < EDIM; ++e) {
            om[e]  = (e == i1 || e == i2) ? 1.0f : 0.0f;
            orp[e] = (e == i1) ? v1 * rs : ((e == i2) ? v2 * rs : 0.0f);
            opf[e] = p[e];
        }
        oi[0] = (float)i1; oi[1] = (float)i2;
    }
}

extern "C" void kernel_launch(void* const* d_in, const int* in_sizes, int n_in,
                              void* d_out, int out_size, void* d_ws, size_t ws_size,
                              hipStream_t stream) {
    const float* inputs = (const float*)d_in[0];
    const float* cond   = (const float*)d_in[1];
    const float* W1     = (const float*)d_in[2];
    const float* b1     = (const float*)d_in[3];
    const float* W2     = (const float*)d_in[4];
    const float* b2     = (const float*)d_in[5];
    float* out = (float*)d_out;

    const size_t needWs = (size_t)(FDIM * DDIM + EDIM * FDIM) * 2;  // bf16 bytes
    int useWs = (d_ws != nullptr && ws_size >= needWs) ? 1 : 0;
    __bf16* W1Tg = (__bf16*)d_ws;
    __bf16* W2Tg = (__bf16*)((char*)d_ws + (size_t)FDIM * DDIM * 2);

    if (useWs) {
        int nconv = FDIM * DDIM + EDIM * FDIM;              // 540672
        convert_weights_kernel<<<(nconv + 255) / 256, 256, 0, stream>>>(W1, W2, W1Tg, W2Tg);
    }
    router_fused_kernel<<<dim3(N_TOK / MT), dim3(256), 0, stream>>>(
        inputs, cond, W1, b1, W2, b2, W1Tg, W2Tg, useWs, out);
}
